// Tagger_22668837388924
// MI455X (gfx1250) — matrix operand order repaired and checked by reading, not compiled, not run on hardware
//
#include <hip/hip_runtime.h>
#include <hip/hip_bf16.h>

// ---------------------------------------------------------------------------
// Bidirectional 2-layer LSTM tagger for MI455X (gfx1250, wave32, WMMA + TDM).
// - All GEMMs via v_wmma_f32_16x16x32_f16 (f32 accumulate).
// - Gates column-padded so all 4 gate values for one (row,col) land in the
//   same lane/acc slot -> cell state lives in VGPRs for the whole sequence.
// - Recurrent weights (Whh) preloaded into VGPRs (loop-invariant).
// - Next step's input tile staged into LDS by the Tensor Data Mover
//   (tensor_load_to_lds, TENSORcnt), double-buffered one step ahead.
// ---------------------------------------------------------------------------

#define S_LEN 512
#define B_SZ  512
#define E_DIM 20

typedef _Float16 half_t;
typedef __attribute__((ext_vector_type(16))) _Float16 v16h;
typedef __attribute__((ext_vector_type(8)))  _Float16 v8h;
typedef __attribute__((ext_vector_type(8)))  float    v8f;
typedef __attribute__((ext_vector_type(4)))  unsigned int u32x4;
typedef __attribute__((ext_vector_type(8)))  int i32x8;
typedef __attribute__((ext_vector_type(4)))  int i32x4;

static __device__ __forceinline__ v8f wmma_f16(v16h a, v16h b, v8f c) {
  return __builtin_amdgcn_wmma_f32_16x16x32_f16(false, a, false, b, (short)0, c,
                                                false, false);
}

static __device__ __forceinline__ v16h combine(v8h lo, v8h hi) {
  v16h o;
#pragma unroll
  for (int i = 0; i < 8; ++i) { o[i] = lo[i]; o[i + 8] = hi[i]; }
  return o;
}

// A-matrix tile (16x32 f16). base -> (row0, k0). lane r = row.
// Lane-half h holds K chunks {h*8 .. h*8+7} and {16+h*8 .. 16+h*8+7}.
static __device__ __forceinline__ v16h load_a(const half_t* base,
                                              size_t row_stride, int lane) {
  const int h = lane >> 4, r = lane & 15;
  const half_t* p = base + (size_t)r * row_stride + h * 8;
  v8h lo = *(const v8h*)p;
  v8h hi = *(const v8h*)(p + 16);
  return combine(lo, hi);
}

// B-matrix tile (32x16 f16) from W stored [n][k] row-major. lane r = column n.
// Lane-half h holds K chunks {h*8 .. h*8+7} and {16+h*8 .. 16+h*8+7}.
static __device__ __forceinline__ v16h load_b(const half_t* base,
                                              size_t row_stride, int lane) {
  const int h = lane >> 4, r = lane & 15;
  const half_t* p = base + (size_t)r * row_stride + h * 8;
  v8h lo = *(const v8h*)p;
  v8h hi = *(const v8h*)(p + 16);
  return combine(lo, hi);
}

static __device__ __forceinline__ float sigmoidf_(float x) {
  return 1.0f / (1.0f + __expf(-x));
}

// ---------------------------------------------------------------------------
// Tensor Data Mover: 2D tile load global -> LDS.
//   tile = tile_h rows x tile_w f16 elements, row pitch = row_stride_elems,
//   packed contiguously into LDS at lds_byte_addr. count=1, data_size=2B,
//   no padding/iteration/multicast, type=2 (image).
// ---------------------------------------------------------------------------
static __device__ __forceinline__ void tdm_load_2d(unsigned lds_byte_addr,
                                                   const void* gptr,
                                                   unsigned tile_w_elems,
                                                   unsigned tile_h_rows,
                                                   unsigned long long row_stride_elems) {
  unsigned long long ga = (unsigned long long)(uintptr_t)gptr;
  u32x4 g0;
  g0[0] = 1u;                                            // count=1 (valid user D#)
  g0[1] = lds_byte_addr;                                 // lds_addr
  g0[2] = (unsigned)(ga & 0xffffffffu);                  // global_addr[31:0]
  g0[3] = (unsigned)((ga >> 32) & 0x01ffffffu)           // global_addr[56:32]
          | (2u << 30);                                  // type=2
  i32x8 g1;
  g1[0] = (int)(1u << 16);                               // data_size=1 -> 2 bytes
  g1[1] = (int)((tile_w_elems & 0xffffu) << 16);         // tensor_dim0[15:0]
  g1[2] = (int)(((tile_w_elems >> 16) & 0xffffu)         // tensor_dim0[31:16]
                | (0xffffu << 16));                      // tensor_dim1[15:0] (big)
  g1[3] = (int)(0x7fffu                                  // tensor_dim1[31:16]
                | ((tile_w_elems & 0xffffu) << 16));     // tile_dim0
  g1[4] = (int)(tile_h_rows & 0xffffu);                  // tile_dim1 (tile_dim2=0)
  g1[5] = (int)(unsigned)(row_stride_elems & 0xffffffffu);        // dim0_stride lo
  g1[6] = (int)(unsigned)((row_stride_elems >> 32) & 0xffffu);    // dim0_stride hi
  g1[7] = 0;                                             // dim1_stride (unused, 2D)
  i32x4 z4 = (i32x4)0;                                   // groups 2/3: unused (<=2D)
#if defined(__clang_major__) && (__clang_major__ >= 23)
  i32x8 z8 = (i32x8)0;
  __builtin_amdgcn_tensor_load_to_lds(g0, g1, z4, z4, z8, 0);
#else
  __builtin_amdgcn_tensor_load_to_lds(g0, g1, z4, z4, 0);
#endif
}

// ---------------------------------------------------------------------------
// Weight packer: src W is (ngates*Hd, K) f32; dst (ngates*HPd, KP) f16, each
// gate's rows padded Hd->HPd, K padded K->KP with zeros.
// ---------------------------------------------------------------------------
__global__ void pack_w_kernel(const float* __restrict__ W, half_t* __restrict__ dst,
                              int ngates, int Hd, int HPd, int K, int KP) {
  size_t total = (size_t)ngates * HPd * KP;
  for (size_t i = (size_t)blockIdx.x * blockDim.x + threadIdx.x; i < total;
       i += (size_t)gridDim.x * blockDim.x) {
    int k  = (int)(i % KP);
    int np = (int)(i / KP);
    int q = np / HPd, j = np % HPd;
    float v = (j < Hd && k < K) ? W[((size_t)q * Hd + j) * K + k] : 0.0f;
    dst[i] = (half_t)v;
  }
}

__global__ void pack_bias_kernel(const float* __restrict__ bih,
                                 const float* __restrict__ bhh,
                                 float* __restrict__ dst, int Hd, int HPd) {
  int i = blockIdx.x * blockDim.x + threadIdx.x;
  if (i >= 4 * HPd) return;
  int q = i / HPd, j = i % HPd;
  dst[i] = (j < Hd) ? (bih[q * Hd + j] + bhh[q * Hd + j]) : 0.0f;
}

// ---------------------------------------------------------------------------
// Embedding gather: X[s][b][0..31] = f16(emb[x[b][s]][0..19]) zero-padded.
// ---------------------------------------------------------------------------
__global__ void embed_gather_kernel(const int* __restrict__ x,
                                    const float* __restrict__ emb_f,
                                    const float* __restrict__ emb_b,
                                    half_t* __restrict__ Xf,
                                    half_t* __restrict__ Xb) {
  size_t idx = (size_t)blockIdx.x * blockDim.x + threadIdx.x;
  const size_t total = (size_t)S_LEN * B_SZ * 32;
  if (idx >= total) return;
  int k = (int)(idx & 31);
  size_t sb = idx >> 5;
  int b = (int)(sb % B_SZ);
  int s = (int)(sb / B_SZ);
  int tok = x[(size_t)b * S_LEN + s];
  float vf = 0.0f, vb = 0.0f;
  if (k < E_DIM) {
    vf = emb_f[(size_t)tok * E_DIM + k];
    vb = emb_b[(size_t)tok * E_DIM + k];
  }
  Xf[idx] = (half_t)vf;
  Xb[idx] = (half_t)vb;
}

// ---------------------------------------------------------------------------
// LSTM recurrence. H = hidden, HP = per-gate padded width (mult of 16),
// KINP = padded input width (mult of 32), ROWS = batch rows per block,
// OUTS = output row stride (halfs). N layout: [i:HP][f:HP][g:HP][o:HP].
// Whh in VGPRs, h via LDS, X double-buffered in LDS by the TDM, c in VGPRs.
// ---------------------------------------------------------------------------
template <int H, int HP, int KINP, int ROWS, int OUTS>
__global__ void __launch_bounds__(256)
lstm_kernel(const half_t* __restrict__ Xseq, size_t x_step_stride, size_t x_row_stride,
            const half_t* __restrict__ Wih,  // [4*HP][KINP] f16 packed
            const half_t* __restrict__ Whh,  // [4*HP][HP]   f16 packed
            const float*  __restrict__ bsum, // [4*HP]
            const float*  __restrict__ h0,   // [B][H]
            const float*  __restrict__ c0,   // [B][H]
            half_t* __restrict__ out,        // [S][B][OUTS]
            int fwd, int out_col0) {
  constexpr int UT    = HP / 16;
  constexpr int NT    = 4 * UT;
  constexpr int KT_IN = KINP / 32;
  constexpr int KT_H  = HP / 32;

  __shared__ __align__(16) half_t hbuf[ROWS * HP];          // hidden state
  __shared__ __align__(16) half_t xbuf[2][ROWS * KINP];     // TDM double buffer

  const int lane = threadIdx.x & 31;
  const int wave = (int)(threadIdx.x >> 5);
  const int hl   = lane >> 4;
  const int r    = lane & 15;
  const int row0 = blockIdx.x * ROWS;
  const int m0   = wave * 16;

  // Recurrent weights are loop-invariant: park them in VGPRs.
  v16h whh[NT][KT_H];
#pragma unroll
  for (int nt = 0; nt < NT; ++nt)
#pragma unroll
    for (int kt = 0; kt < KT_H; ++kt)
      whh[nt][kt] = load_b(Whh + (size_t)(nt * 16) * HP + kt * 32, HP, lane);

  // Init h(0) in LDS (zero-pad cols H..HP-1 -> padded K columns contribute 0).
  for (int idx = threadIdx.x; idx < ROWS * HP; idx += blockDim.x) {
    int mm = idx / HP, jj = idx % HP;
    float v = (jj < H) ? h0[(size_t)(row0 + mm) * H + jj] : 0.0f;
    hbuf[idx] = (half_t)v;
  }

  // Cell state + fused biases in registers for the whole sequence.
  float creg[UT][8];
  float bias[4][UT];
#pragma unroll
  for (int u = 0; u < UT; ++u) {
    const int j = u * 16 + r;
#pragma unroll
    for (int jj = 0; jj < 8; ++jj) {
      const int m = row0 + m0 + hl * 8 + jj;
      creg[u][jj] = (j < H) ? c0[(size_t)m * H + j] : 0.0f;
    }
#pragma unroll
    for (int q = 0; q < 4; ++q) bias[q][u] = bsum[q * HP + j];
  }

  // TDM prologue: stage X(step 0) into xbuf[0].
  if (wave == 0) {
    const int ts0 = fwd ? 0 : (S_LEN - 1);
    tdm_load_2d((unsigned)(uintptr_t)&xbuf[0][0],
                Xseq + (size_t)ts0 * x_step_stride + (size_t)row0 * x_row_stride,
                KINP, ROWS, (unsigned long long)x_row_stride);
  }
  __syncthreads();  // hbuf init done (xbuf[0] published by the in-loop wait)

  for (int t = 0; t < S_LEN; ++t) {
    const int ts = fwd ? t : (S_LEN - 1 - t);

    // Issue next step's X tile into the other buffer; then drain the current
    // one (TDM completes in-order: tensorcnt<=1 => load t is finished).
    if (wave == 0) {
      if (t + 1 < S_LEN) {
        const int tsn = fwd ? (t + 1) : (S_LEN - 2 - t);
        tdm_load_2d((unsigned)(uintptr_t)&xbuf[(t + 1) & 1][0],
                    Xseq + (size_t)tsn * x_step_stride + (size_t)row0 * x_row_stride,
                    KINP, ROWS, (unsigned long long)x_row_stride);
        __builtin_amdgcn_s_wait_tensorcnt((short)1);
      } else {
        __builtin_amdgcn_s_wait_tensorcnt((short)0);
      }
    }
    __syncthreads();  // xbuf[t&1] ready; h(t) (written last iter) visible

    const half_t* Xt = &xbuf[t & 1][0] + (size_t)m0 * KINP;

    v16h ax[KT_IN];
#pragma unroll
    for (int kt = 0; kt < KT_IN; ++kt)
      ax[kt] = load_a(Xt + kt * 32, KINP, lane);

    v16h ah[KT_H];
#pragma unroll
    for (int kt = 0; kt < KT_H; ++kt)
      ah[kt] = load_a(hbuf + (size_t)m0 * HP + kt * 32, HP, lane);

    v8f acc[NT];
#pragma unroll
    for (int nt = 0; nt < NT; ++nt) {
      v8f c;
#pragma unroll
      for (int i = 0; i < 8; ++i) c[i] = 0.0f;
#pragma unroll
      for (int kt = 0; kt < KT_IN; ++kt)
        c = wmma_f16(ax[kt],
                     load_b(Wih + (size_t)(nt * 16) * KINP + kt * 32, KINP, lane), c);
#pragma unroll
      for (int kt = 0; kt < KT_H; ++kt)
        c = wmma_f16(ah[kt], whh[nt][kt], c);
      acc[nt] = c;
    }

    __syncthreads();  // all waves done reading h(t) and xbuf[t&1]

    const int out_t = fwd ? ts : (S_LEN - 1 - ts);
#pragma unroll
    for (int u = 0; u < UT; ++u) {
      const int j = u * 16 + r;  // gate-local column
#pragma unroll
      for (int jj = 0; jj < 8; ++jj) {
        const int mloc = m0 + hl * 8 + jj;
        float iv = acc[0 * UT + u][jj] + bias[0][u];
        float fv = acc[1 * UT + u][jj] + bias[1][u];
        float gv = acc[2 * UT + u][jj] + bias[2][u];
        float ov = acc[3 * UT + u][jj] + bias[3][u];
        float cn = sigmoidf_(fv) * creg[u][jj] + sigmoidf_(iv) * tanhf(gv);
        float hn = sigmoidf_(ov) * tanhf(cn);
        creg[u][jj] = cn;
        if (j < H) {
          half_t hh = (half_t)hn;
          hbuf[mloc * HP + j] = hh;  // padded cols stay zero forever
          out[((size_t)out_t * B_SZ + (row0 + mloc)) * OUTS + out_col0 + j] = hh;
        }
      }
    }
    // next iteration's top barrier publishes h(t+1) and the next X buffer
  }
}

// ---------------------------------------------------------------------------
// Final linear: (S*B, 100->128) x (45->48, 100->128)^T + b, f32 out.
// ---------------------------------------------------------------------------
__global__ void __launch_bounds__(256)
final_gemm_kernel(const half_t* __restrict__ l2,    // [S*B][128]
                  const half_t* __restrict__ Wl,    // [48][128]
                  const float*  __restrict__ lin_b, // [45]
                  float* __restrict__ out) {        // [S*B][45]
  const size_t SB = (size_t)S_LEN * B_SZ;
  const int lane = threadIdx.x & 31;
  const size_t wv = ((size_t)blockIdx.x * blockDim.x + threadIdx.x) >> 5;
  const size_t m0 = wv * 16;
  if (m0 >= SB) return;

  v8f acc[3];
#pragma unroll
  for (int nt = 0; nt < 3; ++nt)
#pragma unroll
    for (int i = 0; i < 8; ++i) acc[nt][i] = 0.0f;

#pragma unroll
  for (int kt = 0; kt < 4; ++kt) {
    v16h a = load_a(l2 + m0 * 128 + kt * 32, 128, lane);
#pragma unroll
    for (int nt = 0; nt < 3; ++nt) {
      v16h b = load_b(Wl + (size_t)(nt * 16) * 128 + kt * 32, 128, lane);
      acc[nt] = wmma_f16(a, b, acc[nt]);
    }
  }

  const int hl = lane >> 4, r = lane & 15;
#pragma unroll
  for (int nt = 0; nt < 3; ++nt) {
    int n = nt * 16 + r;
    if (n < 45) {
      float bb = lin_b[n];
#pragma unroll
      for (int jj = 0; jj < 8; ++jj) {
        size_t m = m0 + (size_t)(hl * 8 + jj);
        out[m * 45 + n] = acc[nt][jj] + bb;
      }
    }
  }
}

// ---------------------------------------------------------------------------
extern "C" void kernel_launch(void* const* d_in, const int* in_sizes, int n_in,
                              void* d_out, int out_size, void* d_ws, size_t ws_size,
                              hipStream_t stream) {
  (void)in_sizes; (void)n_in; (void)out_size; (void)ws_size;

  const int*   x      = (const int*)d_in[0];
  const float* emb_f1 = (const float*)d_in[1];
  const float* emb_b1 = (const float*)d_in[2];
  const float* h0_f1  = (const float*)d_in[3];
  const float* c0_f1  = (const float*)d_in[4];
  const float* Wih_f1 = (const float*)d_in[5];
  const float* Whh_f1 = (const float*)d_in[6];
  const float* bih_f1 = (const float*)d_in[7];
  const float* bhh_f1 = (const float*)d_in[8];
  const float* h0_b1  = (const float*)d_in[9];
  const float* c0_b1  = (const float*)d_in[10];
  const float* Wih_b1 = (const float*)d_in[11];
  const float* Whh_b1 = (const float*)d_in[12];
  const float* bih_b1 = (const float*)d_in[13];
  const float* bhh_b1 = (const float*)d_in[14];
  const float* h0_f2  = (const float*)d_in[15];
  const float* c0_f2  = (const float*)d_in[16];
  const float* Wih_f2 = (const float*)d_in[17];
  const float* Whh_f2 = (const float*)d_in[18];
  const float* bih_f2 = (const float*)d_in[19];
  const float* bhh_f2 = (const float*)d_in[20];
  const float* h0_b2  = (const float*)d_in[21];
  const float* c0_b2  = (const float*)d_in[22];
  const float* Wih_b2 = (const float*)d_in[23];
  const float* Whh_b2 = (const float*)d_in[24];
  const float* bih_b2 = (const float*)d_in[25];
  const float* bhh_b2 = (const float*)d_in[26];
  const float* lin_w  = (const float*)d_in[27];
  const float* lin_b  = (const float*)d_in[28];
  float* out = (float*)d_out;

  // Bump allocator over d_ws (~134 MB total; fits in the 192 MB L2).
  char* ws = (char*)d_ws;
  size_t off = 0;
  auto alloc = [&](size_t bytes) -> char* {
    char* p = ws + off;
    off += bytes;
    off = (off + 255) & ~(size_t)255;
    return p;
  };

  const size_t SB = (size_t)S_LEN * B_SZ;
  half_t* Xf1 = (half_t*)alloc(SB * 32 * sizeof(half_t));
  half_t* Xb1 = (half_t*)alloc(SB * 32 * sizeof(half_t));
  half_t* l1  = (half_t*)alloc(SB * 64 * sizeof(half_t));   // [t*B+b][64]
  half_t* l2  = (half_t*)alloc(SB * 128 * sizeof(half_t));  // [t*B+b][128]
  half_t* Wihf1p = (half_t*)alloc(128 * 32 * sizeof(half_t));
  half_t* Whhf1p = (half_t*)alloc(128 * 32 * sizeof(half_t));
  half_t* Wihb1p = (half_t*)alloc(128 * 32 * sizeof(half_t));
  half_t* Whhb1p = (half_t*)alloc(128 * 32 * sizeof(half_t));
  half_t* Wihf2p = (half_t*)alloc(256 * 64 * sizeof(half_t));
  half_t* Whhf2p = (half_t*)alloc(256 * 64 * sizeof(half_t));
  half_t* Wihb2p = (half_t*)alloc(256 * 64 * sizeof(half_t));
  half_t* Whhb2p = (half_t*)alloc(256 * 64 * sizeof(half_t));
  half_t* Wlp    = (half_t*)alloc(48 * 128 * sizeof(half_t));
  float* bsf1 = (float*)alloc(128 * sizeof(float));
  float* bsb1 = (float*)alloc(128 * sizeof(float));
  float* bsf2 = (float*)alloc(256 * sizeof(float));
  float* bsb2 = (float*)alloc(256 * sizeof(float));

  // --- pack weights into gate-padded f16 [n][k] layouts ---
  pack_w_kernel<<<64, 256, 0, stream>>>(Wih_f1, Wihf1p, 4, 30, 32, 20, 32);
  pack_w_kernel<<<64, 256, 0, stream>>>(Whh_f1, Whhf1p, 4, 30, 32, 30, 32);
  pack_w_kernel<<<64, 256, 0, stream>>>(Wih_b1, Wihb1p, 4, 30, 32, 20, 32);
  pack_w_kernel<<<64, 256, 0, stream>>>(Whh_b1, Whhb1p, 4, 30, 32, 30, 32);
  pack_w_kernel<<<64, 256, 0, stream>>>(Wih_f2, Wihf2p, 4, 50, 64, 60, 64);
  pack_w_kernel<<<64, 256, 0, stream>>>(Whh_f2, Whhf2p, 4, 50, 64, 50, 64);
  pack_w_kernel<<<64, 256, 0, stream>>>(Wih_b2, Wihb2p, 4, 50, 64, 60, 64);
  pack_w_kernel<<<64, 256, 0, stream>>>(Whh_b2, Whhb2p, 4, 50, 64, 50, 64);
  pack_w_kernel<<<64, 256, 0, stream>>>(lin_w, Wlp, 1, 45, 48, 100, 128);
  pack_bias_kernel<<<4, 256, 0, stream>>>(bih_f1, bhh_f1, bsf1, 30, 32);
  pack_bias_kernel<<<4, 256, 0, stream>>>(bih_b1, bhh_b1, bsb1, 30, 32);
  pack_bias_kernel<<<4, 256, 0, stream>>>(bih_f2, bhh_f2, bsf2, 50, 64);
  pack_bias_kernel<<<4, 256, 0, stream>>>(bih_b2, bhh_b2, bsb2, 50, 64);

  // --- embedding gather (f32 -> padded f16 step-major [s][b][32]) ---
  embed_gather_kernel<<<(int)(SB * 32 / 256), 256, 0, stream>>>(
      x, emb_f1, emb_b1, Xf1, Xb1);

  // zero pad columns of the inter-layer buffers (cols never written below)
  hipMemsetAsync(l1, 0, SB * 64 * sizeof(half_t), stream);
  hipMemsetAsync(l2, 0, SB * 128 * sizeof(half_t), stream);

  // --- layer 1: fwd and bwd (independent; disjoint output columns) ---
  lstm_kernel<30, 32, 32, 128, 64><<<dim3(B_SZ / 128), 256, 0, stream>>>(
      Xf1, (size_t)B_SZ * 32, 32, Wihf1p, Whhf1p, bsf1, h0_f1, c0_f1, l1, 1, 0);
  lstm_kernel<30, 32, 32, 128, 64><<<dim3(B_SZ / 128), 256, 0, stream>>>(
      Xb1, (size_t)B_SZ * 32, 32, Wihb1p, Whhb1p, bsb1, h0_b1, c0_b1, l1, 0, 30);

  // --- layer 2: reads l1 through the reference's (S,B)->(B,S) view quirk:
  // step t2 gathers flat rows {b*S + t2}: base += t2*64, row stride S*64.
  lstm_kernel<50, 64, 64, 128, 128><<<dim3(B_SZ / 128), 256, 0, stream>>>(
      l1, (size_t)64, (size_t)S_LEN * 64, Wihf2p, Whhf2p, bsf2, h0_f2, c0_f2,
      l2, 1, 0);
  lstm_kernel<50, 64, 64, 128, 128><<<dim3(B_SZ / 128), 256, 0, stream>>>(
      l1, (size_t)64, (size_t)S_LEN * 64, Wihb2p, Whhb2p, bsb2, h0_b2, c0_b2,
      l2, 0, 50);

  // --- final linear projection to tags ---
  final_gemm_kernel<<<(int)(SB / 16 / 8), 256, 0, stream>>>(l2, Wlp, lin_b, out);
}
